// SpMiddlePillarEncoderNeckMergeF_51015621542416
// MI455X (gfx1250) — matrix operand order repaired and checked by reading, not compiled, not run on hardware
//
#include <hip/hip_runtime.h>
#include <math.h>

// ---------------------------------------------------------------------------
// SpMiddlePillarEncoderNeckMerge for MI455X (gfx1250, wave32, WMMA)
//
// Assumed d_in layout (setup_inputs dict order; params pytree flattened in
// JAX key-sorted DFS order, cb/mlp leaves sorted b,g,w):
//   0 xyz (NPTS*3 f32), 1 xyz_batch_cnt (2 i32), 2 pt_features (NPTS*2 f32)
//   3.. params leaves (see index constants in kernel_launch)
// d_out: (2, 512, 64, 64) f32 NCHW.
// ---------------------------------------------------------------------------

typedef __attribute__((ext_vector_type(16))) __bf16 v16bf;
typedef __attribute__((ext_vector_type(8)))  __bf16 v8bf;
typedef __attribute__((ext_vector_type(8)))  float  v8f;

#define BATCH    2
#define PCMIN    (-51.2f)

// ------------------------------- utility kernels ---------------------------

__global__ void zero_f32_kernel(float* p, long n) {
  long i = (long)blockIdx.x * blockDim.x + threadIdx.x;
  if (i < n) p[i] = 0.0f;
}

__global__ void cvt_bf16_kernel(const float* __restrict__ s, __bf16* __restrict__ d, long n) {
  long i = (long)blockIdx.x * blockDim.x + threadIdx.x;
  if (i < n) d[i] = (__bf16)s[i];
}

// fp32 HWIO weights -> bf16 WMMA B-fragment-packed layout:
//   dst[((kc*(COUT/16) + nt)*32 + lane)*16 + e] =
//       src[(kc*32 + (lane>>4)*16 + e)*COUT + nt*16 + (lane&15)]
// so each lane's 16-element B fragment is one contiguous 32-byte load.
__global__ void pack_wt_kernel(const float* __restrict__ src, __bf16* __restrict__ dst,
                               int K, int COUT) {
  long i = (long)blockIdx.x * blockDim.x + threadIdx.x;
  long tot = (long)K * COUT;
  if (i >= tot) return;
  int  e    = (int)(i & 15);
  int  lane = (int)((i >> 4) & 31);
  long blk  = i >> 9;                    // kc*(COUT/16) + nt
  int  numNT = COUT >> 4;
  int  nt = (int)(blk % numNT);
  int  kc = (int)(blk / numNT);
  int  k  = kc * 32 + (lane >> 4) * 16 + e;
  int  n  = nt * 16 + (lane & 15);
  dst[i] = (__bf16)src[(long)k * COUT + n];
}

// point -> pillar scatter-max with pointwise MLP (PillarMaxPooling)
__global__ void pillar_pool_kernel(const float* __restrict__ xyz,
                                   const float* __restrict__ ptf,
                                   const int*   __restrict__ bcnt,
                                   const float* __restrict__ W,   // (8, C)
                                   const float* __restrict__ Gm,  // (C)
                                   const float* __restrict__ Bt,  // (C)
                                   float bev, int G, int C,
                                   float* __restrict__ grid,      // (B,G,G,C) zero-init
                                   float* __restrict__ mask,      // (B,G,G)  zero-init
                                   int npts) {
  int i = blockIdx.x * blockDim.x + threadIdx.x;
  if (i >= npts) return;
  int b = (i < bcnt[0]) ? 0 : 1;
  float x = xyz[i * 3 + 0], y = xyz[i * 3 + 1], z = xyz[i * 3 + 2];
  int ix = (int)((x - PCMIN) / bev); ix = min(max(ix, 0), G - 1);
  int iy = (int)((y - PCMIN) / bev); iy = min(max(iy, 0), G - 1);
  float cx = PCMIN + ((float)ix + 0.5f) * bev;
  float cy = PCMIN + ((float)iy + 0.5f) * bev;
  float aug[8] = { x, y, z, x - cx, y - cy, sqrtf(x * x + y * y + z * z),
                   ptf[i * 2 + 0], ptf[i * 2 + 1] };
  long seg = ((long)b * G + iy) * G + ix;
  float* cell = grid + seg * C;
  for (int j = 0; j < C; ++j) {
    float acc = 0.0f;
#pragma unroll
    for (int t = 0; t < 8; ++t) acc += aug[t] * W[t * C + j];
    float v = fmaxf(acc * Gm[j] + Bt[j], 0.0f);       // relu -> v >= 0
    atomicMax((int*)&cell[j], __float_as_int(v));     // int-max == float-max for >=0
  }
  mask[seg] = 1.0f;
}

// occupancy dilation: 3x3 stride-2 max-pool, pad 1
__global__ void ds_mask_kernel(const float* __restrict__ src, float* __restrict__ dst,
                               int Hin, int Win, int Hout, int Wout) {
  int i = blockIdx.x * blockDim.x + threadIdx.x;
  int tot = BATCH * Hout * Wout;
  if (i >= tot) return;
  int x = i % Wout, y = (i / Wout) % Hout, b = i / (Wout * Hout);
  float mv = 0.0f;
  for (int r = 0; r < 3; ++r) {
    int yy = 2 * y - 1 + r; if (yy < 0 || yy >= Hin) continue;
    for (int c = 0; c < 3; ++c) {
      int xx = 2 * x - 1 + c; if (xx < 0 || xx >= Win) continue;
      mv = fmaxf(mv, src[((long)b * Hin + yy) * Win + xx]);
    }
  }
  dst[i] = (mv > 0.0f) ? 1.0f : 0.0f;
}

// y = x + (m4>0 ? pool4 : 0), bf16 out
__global__ void add_pool4_kernel(const __bf16* __restrict__ x, const float* __restrict__ pool4,
                                 const float* __restrict__ m4, __bf16* __restrict__ yo,
                                 long n, int C) {
  long i = (long)blockIdx.x * blockDim.x + threadIdx.x;
  if (i >= n) return;
  long pix = i / C;
  float v = (float)x[i] + ((m4[pix] > 0.0f) ? pool4[i] : 0.0f);
  yo[i] = (__bf16)v;
}

// concat(neck4, neck5) NHWC bf16 -> NCHW f32
__global__ void pack_out_kernel(const __bf16* __restrict__ n4, const __bf16* __restrict__ n5,
                                float* __restrict__ out) {
  long i = (long)blockIdx.x * blockDim.x + threadIdx.x;
  long tot = (long)BATCH * 512 * 64 * 64;
  if (i >= tot) return;
  int x  = (int)(i % 64);
  int y  = (int)((i / 64) % 64);
  int ch = (int)((i / 4096) % 512);
  int b  = (int)(i / (4096L * 512));
  long src = (((long)b * 64 + y) * 64 + x) * 256;
  out[i] = (ch < 256) ? (float)n4[src + ch] : (float)n5[src + ch - 256];
}

// ------------------------- WMMA implicit-GEMM 3x3 conv ---------------------
// SM = 1 (stride 1), 2 (stride 2), 0 (2x transposed conv, pad (1,2))
// Block: 256 threads (8 wave32) covering a 32-pixel output row (2 M tiles).
// Each wave owns one 16-cout tile and issues TWO wmma per K-chunk (M tiles
// share the B fragment -> B global traffic amortized 2x). A fragments are
// ds_load_b128 pairs at constant offsets; B is one contiguous 32B global load
// from the fragment-packed weight stream with global_prefetch_b8 lookahead.

template <int CIN, int SM>
__global__ __launch_bounds__(256)
void conv3x3_wmma(const __bf16* __restrict__ in, const __bf16* __restrict__ wt,
                  const float* __restrict__ gam, const float* __restrict__ bet,
                  const __bf16* __restrict__ res, const float* __restrict__ mask,
                  __bf16* __restrict__ out,
                  int Hin, int Win, int Hout, int Wout, int COUT, int doRelu) {
  constexpr bool TR = (SM == 0);
  constexpr int  S  = TR ? 1 : SM;
  constexpr int  PW = TR ? 34 : (32 * S + 2);   // 32-pixel span + halo
  constexpr int  C8 = CIN / 8;
  extern __shared__ __bf16 smem[];  // [3][PW][CIN]

  const int tilesX = Wout >> 5;
  int xt  = blockIdx.x % tilesX;
  int rem = blockIdx.x / tilesX;
  int y   = rem % Hout;
  int b   = rem / Hout;
  int x0  = xt << 5;
  int tid = threadIdx.x;

  // ---- stage zero-padded input patch into LDS, 16B (8 x bf16) granules ----
  const int total8 = 3 * PW * C8;
  for (int t = tid; t < total8; t += 256) {
    int cin8 = t % C8;
    int c    = (t / C8) % PW;
    int r    = t / (C8 * PW);
    v8bf v = {};
    if (!TR) {
      int iy = y * S - 1 + r;
      int ix = x0 * S - 1 + c;
      if (iy >= 0 && iy < Hin && ix >= 0 && ix < Win)
        v = *(const v8bf*)(in + (((long)b * Hin + iy) * Win + ix) * CIN + cin8 * 8);
    } else {
      int ty = y - 1 + r, tx = x0 - 1 + c;
      if (ty >= 0 && tx >= 0 && !(ty & 1) && !(tx & 1)) {
        int iy = ty >> 1, ix = tx >> 1;
        if (iy < Hin && ix < Win)
          v = *(const v8bf*)(in + (((long)b * Hin + iy) * Win + ix) * CIN + cin8 * 8);
      }
    }
    *(v8bf*)(smem + (long)t * 8) = v;
  }
  __syncthreads();

  int wid  = tid >> 5, lane = tid & 31;
  int coutBase = (blockIdx.y * 8 + wid) << 4;
  if (coutBase >= COUT) return;
  int m    = lane & 15;        // N index for B/C/D; M row group select via kgrp
  int kgrp = lane >> 4;
  int cout = coutBase + m;

  // fragment-packed B stream: one v16bf (32B) per lane per K-chunk
  const long  bstride = (long)(COUT >> 4) * 32;          // fragments per K-chunk
  const v16bf* bfrag = (const v16bf*)wt + ((long)(coutBase >> 4) * 32 + lane);

  v8f acc0 = {}, acc1 = {};
  const int NCH = (9 * CIN) / 32;
  for (int kc = 0; kc < NCH; ++kc) {
    int k0      = kc * 32;
    int t9      = k0 / CIN;          // compile-time shift (CIN pow2 >= 32)
    int dy      = t9 / 3, dx = t9 % 3;
    int cinBase = k0 & (CIN - 1);
    // A fragments: contiguous K order (lanes 0-15: K 0-15; 16-31: K 16-31), matching packed B
    const __bf16* arow = smem + ((dy * PW + m * S + dx) * CIN + cinBase + kgrp * 16);
    v16bf a0, a1;
#pragma unroll
    for (int e = 0; e < 8; ++e)  a0[e] = arow[e];
#pragma unroll
    for (int e = 8; e < 16; ++e) a0[e] = arow[e];
    const __bf16* arow1 = arow + 16 * S * CIN;           // second M tile (+16 px)
#pragma unroll
    for (int e = 0; e < 8; ++e)  a1[e] = arow1[e];
#pragma unroll
    for (int e = 8; e < 16; ++e) a1[e] = arow1[e];
    // B fragment: single contiguous 32-byte load, shared by both M tiles
    v16bf bb = *bfrag;
    if (kc + 1 < NCH)
      __builtin_prefetch(bfrag + bstride, 0, 1);         // global_prefetch_b8
    bfrag += bstride;
    acc0 = __builtin_amdgcn_wmma_f32_16x16x32_bf16(false, a0, false, bb,
                                                   (short)0, acc0, false, false);
    acc1 = __builtin_amdgcn_wmma_f32_16x16x32_bf16(false, a1, false, bb,
                                                   (short)0, acc1, false, false);
  }

  // ---- epilogue: BN affine, optional residual, ReLU, occupancy mask ----
  float g  = gam[cout], be = bet[cout];
  long pixBase = ((long)b * Hout + y) * Wout + x0;
  const float* mrow = mask + pixBase;
  long obase = pixBase * (long)COUT + cout;
#pragma unroll
  for (int r = 0; r < 8; ++r) {
    int mr = r + kgrp * 8;                      // C/D layout: M = r + 8*(lane>=16)
    float v = acc0[r] * g + be;
    if (res) v += (float)res[obase + (long)mr * COUT];
    if (doRelu) v = fmaxf(v, 0.0f);
    v *= mrow[mr];
    out[obase + (long)mr * COUT] = (__bf16)v;

    int mr1 = mr + 16;                          // second M tile
    float v1 = acc1[r] * g + be;
    if (res) v1 += (float)res[obase + (long)mr1 * COUT];
    if (doRelu) v1 = fmaxf(v1, 0.0f);
    v1 *= mrow[mr1];
    out[obase + (long)mr1 * COUT] = (__bf16)v1;
  }
}

// --------------------------------- launcher --------------------------------

extern "C" void kernel_launch(void* const* d_in, const int* in_sizes, int n_in,
                              void* d_out, int out_size, void* d_ws, size_t ws_size,
                              hipStream_t stream) {
  (void)in_sizes; (void)n_in; (void)out_size; (void)ws_size;
  const float* xyz  = (const float*)d_in[0];
  const int*   bcnt = (const int*)d_in[1];
  const float* ptf  = (const float*)d_in[2];
  auto F = [&](int i) { return (const float*)d_in[i]; };
  const int NPT = 240000;

  // bump allocator over d_ws
  size_t off = 0;
  auto alloc = [&](size_t bytes) -> void* {
    off = (off + 255) & ~(size_t)255;
    void* p = (char*)d_ws + off;
    off += bytes;
    return p;
  };
  auto wpack = [&](int idx, int CINv, int COUTv) -> const __bf16* {
    long n = 9L * CINv * COUTv;
    __bf16* p = (__bf16*)alloc((size_t)n * 2);
    pack_wt_kernel<<<(unsigned)((n + 255) / 256), 256, 0, stream>>>(F(idx), p, 9 * CINv, COUTv);
    return p;
  };

  // ---- pack conv weights to bf16 WMMA fragment layout ----
  const __bf16* w_c1c0  = wpack(17, 32, 32);
  const __bf16* w_c1b1a = wpack(5,  32, 32);
  const __bf16* w_c1b1b = wpack(8,  32, 32);
  const __bf16* w_c1b2a = wpack(11, 32, 32);
  const __bf16* w_c1b2b = wpack(14, 32, 32);
  const __bf16* w_ds2   = wpack(71, 32, 64);
  const __bf16* w_c2b1a = wpack(20, 64, 64);
  const __bf16* w_c2b1b = wpack(23, 64, 64);
  const __bf16* w_c2b2a = wpack(26, 64, 64);
  const __bf16* w_c2b2b = wpack(29, 64, 64);
  const __bf16* w_ds3   = wpack(74, 64, 128);
  const __bf16* w_c3b1a = wpack(32, 128, 128);
  const __bf16* w_c3b1b = wpack(35, 128, 128);
  const __bf16* w_c3b2a = wpack(38, 128, 128);
  const __bf16* w_c3b2b = wpack(41, 128, 128);
  const __bf16* w_ds4   = wpack(77, 128, 256);
  const __bf16* w_mrg4  = wpack(86, 256, 256);
  const __bf16* w_c4b1a = wpack(44, 256, 256);
  const __bf16* w_c4b1b = wpack(47, 256, 256);
  const __bf16* w_c4b2a = wpack(50, 256, 256);
  const __bf16* w_c4b2b = wpack(53, 256, 256);
  const __bf16* w_c4up  = wpack(56, 256, 256);
  const __bf16* w_ds5   = wpack(80, 256, 512);
  const __bf16* w_c5b1a = wpack(59, 512, 512);
  const __bf16* w_c5b1b = wpack(62, 512, 512);
  const __bf16* w_c5b2a = wpack(65, 512, 512);
  const __bf16* w_c5b2b = wpack(68, 512, 512);
  const __bf16* w_inv5  = wpack(83, 512, 256);

  // ---- activation / scratch buffers ----
  const size_t ACT = (size_t)2 * 512 * 512 * 32 * 2;  // max bf16 activation bytes
  __bf16* P  = (__bf16*)alloc(ACT);
  char*   QR = (char*)alloc(ACT * 2);
  __bf16* Q  = (__bf16*)QR;
  __bf16* R  = (__bf16*)(QR + ACT);
  float*  pool1f = (float*)QR;  // fp32 scatter grid overlays Q+R (dead before Q use)
  float*  pool4f = (float*)alloc((size_t)2 * 64 * 64 * 256 * 4);
  __bf16* neck4  = (__bf16*)alloc((size_t)2 * 64 * 64 * 256 * 2);
  __bf16* neck5  = (__bf16*)alloc((size_t)2 * 64 * 64 * 256 * 2);
  float*  m1  = (float*)alloc((size_t)2 * 512 * 512 * 4);
  float*  m2  = (float*)alloc((size_t)2 * 256 * 256 * 4);
  float*  m3  = (float*)alloc((size_t)2 * 128 * 128 * 4);
  float*  m4  = (float*)alloc((size_t)2 * 64 * 64 * 4);
  float*  m5  = (float*)alloc((size_t)2 * 32 * 32 * 4);
  float*  mp4 = (float*)alloc((size_t)2 * 64 * 64 * 4);  // scratch pool4 mask

  auto zero = [&](float* p, long n) {
    zero_f32_kernel<<<(unsigned)((n + 255) / 256), 256, 0, stream>>>(p, n);
  };

#define LAUNCH_CONV(CIN_, SM_, inp, wtp, gi, bi, resp, mk, outp, Hin_, Win_, Hout_, Wout_, COUT_, RELU_) \
  do {                                                                                                   \
    constexpr int PW_ = ((SM_) == 0) ? 34 : (32 * (SM_) + 2);                                            \
    dim3 gr_(((Wout_) / 32) * (Hout_)*2, (((COUT_) / 16) + 7) / 8);                                      \
    conv3x3_wmma<CIN_, SM_><<<gr_, 256, (size_t)3 * PW_ * (CIN_)*2, stream>>>(                           \
        inp, wtp, F(gi), F(bi), resp, mk, outp, Hin_, Win_, Hout_, Wout_, COUT_, RELU_);                 \
  } while (0)

  const __bf16* NORES = nullptr;

  // ---- stage 1: pillar pool (G=512, C=32) + conv1 ----
  zero(pool1f, 2L * 512 * 512 * 32);
  zero(m1, 2L * 512 * 512);
  pillar_pool_kernel<<<(NPT + 255) / 256, 256, 0, stream>>>(
      xyz, ptf, bcnt, F(89), F(88), F(87), 0.2f, 512, 32, pool1f, m1, NPT);
  cvt_bf16_kernel<<<(unsigned)((2L * 512 * 512 * 32 + 255) / 256), 256, 0, stream>>>(
      pool1f, P, 2L * 512 * 512 * 32);

  LAUNCH_CONV(32, 1, P, w_c1c0,  16, 15, NORES, m1, Q, 512, 512, 512, 512, 32, 1);
  LAUNCH_CONV(32, 1, Q, w_c1b1a,  4,  3, NORES, m1, R, 512, 512, 512, 512, 32, 1);
  LAUNCH_CONV(32, 1, R, w_c1b1b,  7,  6, Q,     m1, Q, 512, 512, 512, 512, 32, 1);
  LAUNCH_CONV(32, 1, Q, w_c1b2a, 10,  9, NORES, m1, R, 512, 512, 512, 512, 32, 1);
  LAUNCH_CONV(32, 1, R, w_c1b2b, 13, 12, Q,     m1, Q, 512, 512, 512, 512, 32, 1);

  // ---- stage 2 ----
  ds_mask_kernel<<<(2 * 256 * 256 + 255) / 256, 256, 0, stream>>>(m1, m2, 512, 512, 256, 256);
  LAUNCH_CONV(32, 2, Q, w_ds2,   70, 69, NORES, m2, P, 512, 512, 256, 256, 64, 1);
  LAUNCH_CONV(64, 1, P, w_c2b1a, 19, 18, NORES, m2, R, 256, 256, 256, 256, 64, 1);
  LAUNCH_CONV(64, 1, R, w_c2b1b, 22, 21, P,     m2, P, 256, 256, 256, 256, 64, 1);
  LAUNCH_CONV(64, 1, P, w_c2b2a, 25, 24, NORES, m2, R, 256, 256, 256, 256, 64, 1);
  LAUNCH_CONV(64, 1, R, w_c2b2b, 28, 27, P,     m2, P, 256, 256, 256, 256, 64, 1);

  // ---- stage 3 ----
  ds_mask_kernel<<<(2 * 128 * 128 + 255) / 256, 256, 0, stream>>>(m2, m3, 256, 256, 128, 128);
  LAUNCH_CONV(64, 2,  P, w_ds3,   73, 72, NORES, m3, Q, 256, 256, 128, 128, 128, 1);
  LAUNCH_CONV(128, 1, Q, w_c3b1a, 31, 30, NORES, m3, R, 128, 128, 128, 128, 128, 1);
  LAUNCH_CONV(128, 1, R, w_c3b1b, 34, 33, Q,     m3, Q, 128, 128, 128, 128, 128, 1);
  LAUNCH_CONV(128, 1, Q, w_c3b2a, 37, 36, NORES, m3, R, 128, 128, 128, 128, 128, 1);
  LAUNCH_CONV(128, 1, R, w_c3b2b, 40, 39, Q,     m3, Q, 128, 128, 128, 128, 128, 1);

  // ---- stage 4: ds4 + pool4 merge + blocks + neck ----
  ds_mask_kernel<<<(2 * 64 * 64 + 255) / 256, 256, 0, stream>>>(m3, m4, 128, 128, 64, 64);
  LAUNCH_CONV(128, 2, Q, w_ds4, 76, 75, NORES, m4, P, 128, 128, 64, 64, 256, 1);

  zero(pool4f, 2L * 64 * 64 * 256);
  pillar_pool_kernel<<<(NPT + 255) / 256, 256, 0, stream>>>(
      xyz, ptf, bcnt, F(92), F(91), F(90), 1.6f, 64, 256, pool4f, mp4, NPT);
  {
    long n = 2L * 64 * 64 * 256;
    add_pool4_kernel<<<(unsigned)((n + 255) / 256), 256, 0, stream>>>(P, pool4f, m4, R, n, 256);
  }
  LAUNCH_CONV(256, 1, R, w_mrg4,  85, 84, R, m4, P, 64, 64, 64, 64, 256, 1);
  LAUNCH_CONV(256, 1, P, w_c4b1a, 43, 42, NORES, m4, Q, 64, 64, 64, 64, 256, 1);
  LAUNCH_CONV(256, 1, Q, w_c4b1b, 46, 45, P,     m4, P, 64, 64, 64, 64, 256, 1);
  LAUNCH_CONV(256, 1, P, w_c4b2a, 49, 48, NORES, m4, Q, 64, 64, 64, 64, 256, 1);
  LAUNCH_CONV(256, 1, Q, w_c4b2b, 52, 51, P,     m4, P, 64, 64, 64, 64, 256, 1);
  LAUNCH_CONV(256, 1, P, w_c4up,  55, 54, NORES, m4, neck4, 64, 64, 64, 64, 256, 1);

  // ---- stage 5 + inverse conv back to stride-8 grid ----
  ds_mask_kernel<<<(2 * 32 * 32 + 255) / 256, 256, 0, stream>>>(m4, m5, 64, 64, 32, 32);
  LAUNCH_CONV(256, 2, P, w_ds5,   79, 78, NORES, m5, Q, 64, 64, 32, 32, 512, 1);
  LAUNCH_CONV(512, 1, Q, w_c5b1a, 58, 57, NORES, m5, R, 32, 32, 32, 32, 512, 1);
  LAUNCH_CONV(512, 1, R, w_c5b1b, 61, 60, Q,     m5, Q, 32, 32, 32, 32, 512, 1);
  LAUNCH_CONV(512, 1, Q, w_c5b2a, 64, 63, NORES, m5, R, 32, 32, 32, 32, 512, 1);
  LAUNCH_CONV(512, 1, R, w_c5b2b, 67, 66, Q,     m5, Q, 32, 32, 32, 32, 512, 1);
  // SparseInverseConv2d (lhs_dilation=2, pad (1,2)) -> 64x64, masked by m4
  LAUNCH_CONV(512, 0, Q, w_inv5,  82, 81, NORES, m4, neck5, 32, 32, 64, 64, 256, 1);

  // ---- pack concat -> NCHW f32 ----
  {
    long n = (long)BATCH * 512 * 64 * 64;
    pack_out_kernel<<<(unsigned)((n + 255) / 256), 256, 0, stream>>>(neck4, neck5, (float*)d_out);
  }
#undef LAUNCH_CONV
}
